// ColumnParallelLinear_fp8_51745765982773
// MI455X (gfx1250) — 8-bit matrix operand order repaired and checked by reading, compiled, not run on hardware
//
#include <hip/hip_runtime.h>
#include <hip/hip_bf16.h>


typedef __attribute__((ext_vector_type(16))) int   v16i;
typedef __attribute__((ext_vector_type(8)))  float v8f;
typedef __attribute__((ext_vector_type(4)))  int   v4i;
typedef __attribute__((ext_vector_type(4)))  float v4f;

#define M_DIM 4096
#define K_DIM 4096
#define O_DIM 8192
#define NKB   (K_DIM / 128)
#define FP8_MAX 448.0f

#if defined(__AMDGCN__) && __has_builtin(__builtin_amdgcn_global_load_async_to_lds_b128)
#define USE_ASYNC 1
#else
#define USE_ASYNC 0
#endif

#define GAS1(p) ((__attribute__((address_space(1))) v4i*)(p))
#define LAS3(p) ((__attribute__((address_space(3))) v4i*)(p))

#if USE_ASYNC
#if __has_builtin(__builtin_amdgcn_s_wait_asynccnt)
#define WAIT_ASYNC(n) __builtin_amdgcn_s_wait_asynccnt(n)
#else
#define WAIT_ASYNC(n) asm volatile("s_wait_asynccnt %0" ::"i"(n))
#endif
#endif

__global__ __launch_bounds__(256) void quant_x_kernel(
    const float* __restrict__ x,
    unsigned int* __restrict__ xq,
    float* __restrict__ xscale)
{
    const int m = blockIdx.x;
    const int t = threadIdx.x;
    const v4f* row = reinterpret_cast<const v4f*>(x + (size_t)m * K_DIM);

    v4f v[4];
    float amax = 0.0f;
#pragma unroll
    for (int i = 0; i < 4; ++i) {
        v[i] = __builtin_nontemporal_load(row + t * 4 + i);
        amax = fmaxf(amax, fmaxf(fmaxf(fabsf(v[i].x), fabsf(v[i].y)),
                                 fmaxf(fabsf(v[i].z), fabsf(v[i].w))));
    }

    __shared__ float red[256];
    __shared__ float s_inv;
    red[t] = amax;
    __syncthreads();
#pragma unroll
    for (int s = 128; s > 0; s >>= 1) {
        if (t < s) red[t] = fmaxf(red[t], red[t + s]);
        __syncthreads();
    }
    if (t == 0) {
        float a = fmaxf(red[0], 1e-12f);
        xscale[m] = a / FP8_MAX;
        s_inv = FP8_MAX / a;
    }
    __syncthreads();
    const float inv = s_inv;

    uint4 outw;
    unsigned int* ow = &outw.x;
#pragma unroll
    for (int i = 0; i < 4; ++i) {
        int p = 0;
        p = __builtin_amdgcn_cvt_pk_fp8_f32(v[i].x * inv, v[i].y * inv, p, false);
        p = __builtin_amdgcn_cvt_pk_fp8_f32(v[i].z * inv, v[i].w * inv, p, true);
        ow[i] = (unsigned int)p;
    }
    reinterpret_cast<uint4*>(xq + (size_t)m * (K_DIM / 4))[t] = outw;
}

__global__ __launch_bounds__(256) void quant_w_kernel(
    const float* __restrict__ w,
    unsigned int* __restrict__ wq)
{
    const size_t idx = (size_t)blockIdx.x * 256 + threadIdx.x;
    v4f v = __builtin_nontemporal_load(reinterpret_cast<const v4f*>(w) + idx);
    int p = 0;
    p = __builtin_amdgcn_cvt_pk_fp8_f32(v.x, v.y, p, false);
    p = __builtin_amdgcn_cvt_pk_fp8_f32(v.z, v.w, p, true);
    wq[idx] = (unsigned int)p;
}

__device__ __forceinline__ void stage_tiles(
    const unsigned char* __restrict__ gA,
    const unsigned char* __restrict__ gB,
    unsigned char* lA, unsigned char* lB, int t)
{
#pragma unroll
    for (int i = 0; i < 4; ++i) {
        const int c   = i * 256 + t;
        const int row = c >> 3;
        const int col = (c & 7) << 4;
#if USE_ASYNC
        __builtin_amdgcn_global_load_async_to_lds_b128(
            GAS1(gA + (size_t)row * K_DIM + col), LAS3(lA + row * 128 + col), 0, 0);
        __builtin_amdgcn_global_load_async_to_lds_b128(
            GAS1(gB + (size_t)row * K_DIM + col), LAS3(lB + row * 128 + col), 0, 0);
#else
        *reinterpret_cast<uint4*>(lA + row * 128 + col) =
            *reinterpret_cast<const uint4*>(gA + (size_t)row * K_DIM + col);
        *reinterpret_cast<uint4*>(lB + row * 128 + col) =
            *reinterpret_cast<const uint4*>(gB + (size_t)row * K_DIM + col);
#endif
    }
}

__global__ __launch_bounds__(256) void gemm_fp8_wmma_kernel(
    const unsigned char* __restrict__ xq,
    const unsigned char* __restrict__ wq,
    const float* __restrict__ wsi,
    const float* __restrict__ xs,
    float* __restrict__ out)
{
    const int t      = threadIdx.x;
    const int lane   = t & 31;
    const int wid    = t >> 5;
    const int l16    = lane & 15;
    const int hi     = lane >> 4;
    const int wave_m = wid >> 1;
    const int wave_n = wid & 1;
    const int m0 = blockIdx.y * 128;
    const int n0 = blockIdx.x * 128;

    __shared__ __align__(16) unsigned char sA[2][128 * 128];
    __shared__ __align__(16) unsigned char sB[2][128 * 128];

    const unsigned char* gA = xq + (size_t)m0 * K_DIM;
    const unsigned char* gB = wq + (size_t)n0 * K_DIM;
    const float* sp = wsi + (size_t)blockIdx.x * NKB;

    float acc[2][4][8];
#pragma unroll
    for (int a = 0; a < 2; ++a)
#pragma unroll
        for (int b = 0; b < 4; ++b)
#pragma unroll
            for (int e = 0; e < 8; ++e) acc[a][b][e] = 0.0f;

    stage_tiles(gA, gB, sA[0], sB[0], t);
    float skb_next = sp[0];

    for (int kb = 0; kb < NKB; ++kb) {
        const int cur = kb & 1;
        const int nxt = cur ^ 1;
        const float skb_f = skb_next;

        if (kb + 1 < NKB) {
            stage_tiles(gA + (kb + 1) * 128, gB + (kb + 1) * 128, sA[nxt], sB[nxt], t);
            skb_next = sp[kb + 1];
#if USE_ASYNC
            WAIT_ASYNC(8);
#endif
        } else {
#if USE_ASYNC
            WAIT_ASYNC(0);
#endif
        }
        __syncthreads();

        const float skb = __uint_as_float(
            __builtin_amdgcn_readfirstlane(__float_as_uint(skb_f)));

        const unsigned char* lA = sA[cur];
        const unsigned char* lB = sB[cur];

        v16i bf[4];
#pragma unroll
        for (int nt = 0; nt < 4; ++nt) {
            const int coln = wave_n * 64 + nt * 16 + l16;
            const unsigned char* bp = lB + coln * 128 + hi * 16;
#pragma unroll
            for (int g = 0; g < 4; ++g) {
                int4 q = *reinterpret_cast<const int4*>(bp + g * 32);
                bf[nt][4 * g + 0] = q.x;
                bf[nt][4 * g + 1] = q.y;
                bf[nt][4 * g + 2] = q.z;
                bf[nt][4 * g + 3] = q.w;
            }
        }

#pragma unroll
        for (int mt = 0; mt < 2; ++mt) {
            const int rowm = wave_m * 32 + mt * 16 + l16;
            const unsigned char* ap = lA + rowm * 128 + hi * 16;
            v16i af;
#pragma unroll
            for (int p = 0; p < 8; ++p) {
                int2 q = *reinterpret_cast<const int2*>(ap + 32 * (p >> 1) + 8 * (p & 1));
                af[2 * p + 0] = q.x;
                af[2 * p + 1] = q.y;
            }
#pragma unroll
            for (int nt = 0; nt < 4; ++nt) {
                v8f zero = {};
                v8f d = __builtin_amdgcn_wmma_f32_16x16x128_fp8_fp8(
                    af, bf[nt], (short)0, zero, false, false);
#pragma unroll
                for (int e = 0; e < 8; ++e) acc[mt][nt][e] += d[e] * skb;
            }
        }
        __syncthreads();
    }

    float xsv[2][8];
#pragma unroll
    for (int mt = 0; mt < 2; ++mt)
#pragma unroll
        for (int e = 0; e < 8; ++e)
            xsv[mt][e] = xs[m0 + wave_m * 32 + mt * 16 + hi * 8 + e];

#pragma unroll
    for (int mt = 0; mt < 2; ++mt) {
#pragma unroll
        for (int nt = 0; nt < 4; ++nt) {
            const int col = n0 + wave_n * 64 + nt * 16 + l16;
#pragma unroll
            for (int e = 0; e < 8; ++e) {
                const int row = m0 + wave_m * 32 + mt * 16 + hi * 8 + e;
                __builtin_nontemporal_store(acc[mt][nt][e] * xsv[mt][e],
                                            out + (size_t)row * O_DIM + col);
            }
        }
    }
}

extern "C" void kernel_launch(void* const* d_in, const int* in_sizes, int n_in,
                              void* d_out, int out_size, void* d_ws, size_t ws_size,
                              hipStream_t stream) {
    const float* x   = (const float*)d_in[0];
    const float* w   = (const float*)d_in[1];
    const float* wsi = (const float*)d_in[2];
    float* out = (float*)d_out;

    unsigned char* ws = (unsigned char*)d_ws;
    unsigned char* wq = ws;
    unsigned char* xq = ws + (size_t)O_DIM * K_DIM;
    float* xs = (float*)(ws + (size_t)O_DIM * K_DIM + (size_t)M_DIM * K_DIM);

    quant_w_kernel<<<(O_DIM * (size_t)K_DIM) / 4 / 256, 256, 0, stream>>>(
        w, (unsigned int*)wq);

    quant_x_kernel<<<M_DIM, 256, 0, stream>>>(x, (unsigned int*)xq, xs);

    gemm_fp8_wmma_kernel<<<dim3(O_DIM / 128, M_DIM / 128), 256, 0, stream>>>(
        xq, wq, wsi, xs, out);
}
